// MultiHeadAttention_22986664968706
// MI455X (gfx1250) — matrix operand order repaired and checked by reading, not compiled, not run on hardware
//
#include <hip/hip_runtime.h>
#include <hip/hip_bf16.h>

// ---------------------------------------------------------------------------
// MultiHeadAttention for MI455X (gfx1250, wave32, WMMA 16x16x32 f16->f32)
// B=8, N=1024, D=4096, H=8, HEAD_DIM=512, E=64
// GEMMs use double-buffered LDS staging of the shared B tile via
// global_load_async_to_lds_b128 (ASYNCcnt), overlapping copy with WMMA.
// B/K/V fragments are batch-loaded ahead of each WMMA quartet so the
// scheduler can use incremental dscnt/loadcnt waits.
// ---------------------------------------------------------------------------

typedef __attribute__((ext_vector_type(16))) _Float16 v16h;
typedef __attribute__((ext_vector_type(8)))  _Float16 v8h;
typedef __attribute__((ext_vector_type(8)))  float    v8f;

static __device__ __forceinline__ v8f v8f_zero() {
  v8f z;
#pragma unroll
  for (int i = 0; i < 8; ++i) z[i] = 0.0f;
  return z;
}

static __device__ __forceinline__ v8f wmma16(v16h a, v16h b, v8f c) {
  // D = A(16x32 f16) * B(32x16 f16) + C(16x16 f32)
  return __builtin_amdgcn_wmma_f32_16x16x32_f16(
      /*neg_a=*/false, a, /*neg_b=*/false, b,
      /*c_mod=*/(short)0, c, /*reuse_a=*/false, /*reuse_b=*/false);
}

// A-fragment (16x32 f16, row-major source, leading dim lda).
// Wave32 layout: lane L holds row (L&15); kb = (L<16)?0:16;
// halves 0..15 -> k = kb+0..15 (16 consecutive halves).
static __device__ __forceinline__ v16h load_A_frag(const _Float16* A, int lda,
                                                   int row0, int k0, int lane) {
  const int r  = lane & 15;
  const int kb = (lane < 16) ? 0 : 16;
  const _Float16* p = A + (size_t)(row0 + r) * lda + k0 + kb;
  v16h a;
  ((v8h*)&a)[0] = *(const v8h*)(p);
  ((v8h*)&a)[1] = *(const v8h*)(p + 8);
  return a;
}

// B-fragment (32x16 f16) loaded from "BT" storage where BT[n][k] = B[k][n].
// Wave32 layout: lane L holds column n = n0+(L&15);
// halves j=0..15 -> k = k0 + ((L<16)?0:16) + j  (16 consecutive halves).
static __device__ __forceinline__ v16h load_Bt_frag(const _Float16* BT, int ldb,
                                                    int k0, int n0, int lane) {
  const int n  = n0 + (lane & 15);
  const int kb = k0 + ((lane < 16) ? 0 : 16);
  const _Float16* p = BT + (size_t)n * ldb + kb;
  return *(const v16h*)p;  // 32B load
}

// Async copy of one 64x32-f16 B tile (BT[col0..col0+63][kk..kk+31]) into LDS.
// 256 x 16B chunks; 128 threads x 2 chunks. ASYNCcnt += 2 per thread.
static __device__ __forceinline__ void stage_B_async(const _Float16* BT, int ldb,
                                                     int col0, int kk,
                                                     _Float16* lds_buf, int tid) {
#pragma unroll
  for (int c = 0; c < 2; ++c) {
    const int chunk = tid * 2 + c;   // 0..255
    const int row   = chunk >> 2;    // 0..63
    const int seg   = chunk & 3;     // 16B segment within the 64B row slice
    const _Float16* src = BT + (size_t)(col0 + row) * ldb + kk + seg * 8;
    const unsigned dst_lds =
        (unsigned)(unsigned long long)(lds_buf + row * 32 + seg * 8);
    asm volatile("global_load_async_to_lds_b128 %0, %1, off"
                 :: "v"(dst_lds),
                    "v"((unsigned long long)src)
                 : "memory");
  }
}

static __device__ __forceinline__ void wait_async_le2() {
  asm volatile("s_wait_asynccnt 0x2" ::: "memory");
}
static __device__ __forceinline__ void wait_async_0() {
  asm volatile("s_wait_asynccnt 0x0" ::: "memory");
}

// ---------------------------------------------------------------------------
// fp32 -> f16 pack
// ---------------------------------------------------------------------------
__global__ void mha_pack_f16(const float* __restrict__ src,
                             _Float16* __restrict__ dst, long long n) {
  long long i = (long long)blockIdx.x * blockDim.x + threadIdx.x;
  const long long stride = (long long)gridDim.x * blockDim.x;
  for (; i < n; i += stride) dst[i] = (_Float16)src[i];
}

// ---------------------------------------------------------------------------
// QK+V projection GEMM: [8192,4096] x [4096,1536] with scatter epilogue.
// Block = 4 waves = 64 rows x 64 cols; B tile staged in LDS (double buffer).
// ---------------------------------------------------------------------------
__global__ __launch_bounds__(128) void mha_proj_qkv(
    const _Float16* __restrict__ xh,    // [8192,4096]
    const _Float16* __restrict__ wcat,  // [1536,4096] = [Wqk ; Wv]
    const float* __restrict__ bqk,      // [1024]
    const float* __restrict__ bv,       // [512]
    _Float16* __restrict__ qh,          // [B,H,N,E]
    _Float16* __restrict__ kh,          // [B,H,N,E]
    _Float16* __restrict__ vth)         // [B,H,E,N]
{
  __shared__ __align__(64) _Float16 bstage[2][64 * 32];

  const int tid  = threadIdx.x;
  const int lane = tid & 31;
  const int wave = tid >> 5;
  const int row0 = (blockIdx.x * 4 + wave) * 16;
  const int col0 = blockIdx.y * 64;

  v8f acc[4];
#pragma unroll
  for (int t = 0; t < 4; ++t) acc[t] = v8f_zero();

  stage_B_async(wcat, 4096, col0, 0, bstage[0], tid);      // ASYNCcnt = 2
  v16h a_cur = load_A_frag(xh, 4096, row0, 0, lane);

  for (int kk = 0; kk < 4096; kk += 32) {
    const int  cur      = (kk >> 5) & 1;
    const bool has_next = (kk + 32) < 4096;
    if (has_next)
      stage_B_async(wcat, 4096, col0, kk + 32, bstage[cur ^ 1], tid);
    if (has_next) wait_async_le2();  // oldest 2 (= current buffer) complete
    else          wait_async_0();
    __syncthreads();                 // everyone's copies for buf[cur] landed

    const int kn = has_next ? kk + 32 : kk;
    const v16h a_nxt = load_A_frag(xh, 4096, row0, kn, lane);

    // batch the 4 B-fragment LDS loads, then the 4 WMMAs
    v16h b[4];
#pragma unroll
    for (int t = 0; t < 4; ++t)
      b[t] = load_Bt_frag(bstage[cur], 32, 0, 16 * t, lane);
#pragma unroll
    for (int t = 0; t < 4; ++t)
      acc[t] = wmma16(a_cur, b[t], acc[t]);

    __syncthreads();                 // reads of buf[cur] done before reuse
    a_cur = a_nxt;
  }

  const int rbase = (lane < 16) ? 0 : 8;
#pragma unroll
  for (int t = 0; t < 4; ++t) {
    const int n = col0 + 16 * t + (lane & 15);
    const float bias = (n < 1024) ? bqk[n] : bv[n - 1024];
#pragma unroll
    for (int r = 0; r < 8; ++r) {
      const int m   = row0 + rbase + r;
      const int bb  = m >> 10;       // batch
      const int tok = m & 1023;      // token
      const _Float16 hv = (_Float16)(acc[t][r] + bias);
      if (n < 1024) {                // qk region: n = h*128 + e*2 + s
        const int hh = n >> 7;
        const int e  = (n >> 1) & 63;
        _Float16* dst = (n & 1) ? kh : qh;
        dst[((size_t)(bb * 8 + hh) * 1024 + tok) * 64 + e] = hv;
      } else {                       // v region: vi = h*64 + e, transposed
        const int vi = n - 1024;
        const int hh = vi >> 6;
        const int e  = vi & 63;
        vth[((size_t)(bb * 8 + hh) * 64 + e) * 1024 + tok] = hv;
      }
    }
  }
}

// ---------------------------------------------------------------------------
// Flash-style attention: one wave = one (b, h, 16-query tile) over 1024 keys.
// softmax-then-/sqrt(512) quirk folds into the final normalization scalar.
// ---------------------------------------------------------------------------
__global__ __launch_bounds__(128) void mha_attn(
    const _Float16* __restrict__ qh,   // [B,H,N,E]
    const _Float16* __restrict__ kh,   // [B,H,N,E]
    const _Float16* __restrict__ vth,  // [B,H,E,N]
    _Float16* __restrict__ aoh)        // [B*N, 512] (h*64+e fastest)
{
  __shared__ __align__(32) _Float16 plds[4][16 * 32];

  const int lane = threadIdx.x & 31;
  const int wave = threadIdx.x >> 5;
  const int tid  = blockIdx.x * 4 + wave;   // 0..4095
  const int bb   = tid >> 9;
  const int hh   = (tid >> 6) & 7;
  const int q0   = (tid & 63) << 4;

  const _Float16* qbh = qh  + (size_t)(bb * 8 + hh) * 1024 * 64;
  const _Float16* kbh = kh  + (size_t)(bb * 8 + hh) * 1024 * 64;
  const _Float16* vbh = vth + (size_t)(bb * 8 + hh) * 64 * 1024;

  const v16h aq0 = load_A_frag(qbh, 64, q0, 0, lane);
  const v16h aq1 = load_A_frag(qbh, 64, q0, 32, lane);

  float rmax[8], rsum[8];
  v8f o[4];
#pragma unroll
  for (int r = 0; r < 8; ++r) { rmax[r] = -3.0e38f; rsum[r] = 0.0f; }
#pragma unroll
  for (int t = 0; t < 4; ++t) o[t] = v8f_zero();

  _Float16* plw = plds[wave];
  const int rbase = (lane < 16) ? 0 : 8;

  for (int kb = 0; kb < 1024; kb += 32) {
    // scores for 16 queries x 32 keys (2 N-tiles, K=64 in 2 steps);
    // batch all 4 K-fragment loads first, then the 4 WMMAs
    v16h bk[4];
    bk[0] = load_Bt_frag(kbh, 64, 0,  kb,      lane);
    bk[1] = load_Bt_frag(kbh, 64, 32, kb,      lane);
    bk[2] = load_Bt_frag(kbh, 64, 0,  kb + 16, lane);
    bk[3] = load_Bt_frag(kbh, 64, 32, kb + 16, lane);
    v8f s0 = v8f_zero(), s1 = v8f_zero();
    s0 = wmma16(aq0, bk[0], s0);
    s0 = wmma16(aq1, bk[1], s0);
    s1 = wmma16(aq0, bk[2], s1);
    s1 = wmma16(aq1, bk[3], s1);

    // prefetch the 4 V fragments for this chunk while softmax runs
    v16h bv[4];
#pragma unroll
    for (int t = 0; t < 4; ++t)
      bv[t] = load_Bt_frag(vbh + kb, 1024, 0, 16 * t, lane);

    // online softmax; row stats live in the 16-lane half owning C rows,
    // so xor-shuffles with mask<16 stay inside the group.
#pragma unroll
    for (int r = 0; r < 8; ++r) {
      float m = fmaxf(s0[r], s1[r]);
#pragma unroll
      for (int off = 1; off < 16; off <<= 1) m = fmaxf(m, __shfl_xor(m, off, 32));
      const float nm   = fmaxf(rmax[r], m);
      const float corr = __expf(rmax[r] - nm);
      rmax[r] = nm;
      const float p0 = __expf(s0[r] - nm);
      const float p1 = __expf(s1[r] - nm);
      float ps = p0 + p1;
#pragma unroll
      for (int off = 1; off < 16; off <<= 1) ps += __shfl_xor(ps, off, 32);
      rsum[r] = rsum[r] * corr + ps;
#pragma unroll
      for (int t = 0; t < 4; ++t) o[t][r] *= corr;
      // stage P (C-layout) into per-wave LDS tile for A-layout reload
      const int row = rbase + r;
      plw[row * 32 + (lane & 15)]      = (_Float16)p0;
      plw[row * 32 + 16 + (lane & 15)] = (_Float16)p1;
    }
    asm volatile("s_wait_dscnt 0" ::: "memory");

    const v16h ap = load_A_frag(plw, 32, 0, 0, lane);
#pragma unroll
    for (int t = 0; t < 4; ++t)
      o[t] = wmma16(ap, bv[t], o[t]);
    // DS ops from one wave stay in order at the LDS unit, so next
    // iteration's stores cannot pass the ap load.
  }

  const float inv_sqrt_hd = 0.04419417382415922f;  // 1/sqrt(512)
#pragma unroll
  for (int t = 0; t < 4; ++t) {
#pragma unroll
    for (int r = 0; r < 8; ++r) {
      const float val = o[t][r] * inv_sqrt_hd / rsum[r];
      const int tok = bb * 1024 + q0 + rbase + r;
      aoh[(size_t)tok * 512 + hh * 64 + 16 * t + (lane & 15)] = (_Float16)val;
    }
  }
}

// ---------------------------------------------------------------------------
// Output projection: [8192,512] x [512,4096] + bias -> fp32
// Same async double-buffered B staging as the input projection.
// ---------------------------------------------------------------------------
__global__ __launch_bounds__(128) void mha_out_proj(
    const _Float16* __restrict__ aoh,  // [8192,512]
    const _Float16* __restrict__ woh,  // [4096,512] (= Wo, BT layout)
    const float* __restrict__ bo,      // [4096]
    float* __restrict__ out)           // [8192,4096]
{
  __shared__ __align__(64) _Float16 bstage[2][64 * 32];

  const int tid  = threadIdx.x;
  const int lane = tid & 31;
  const int wave = tid >> 5;
  const int row0 = (blockIdx.x * 4 + wave) * 16;
  const int col0 = blockIdx.y * 64;

  v8f acc[4];
#pragma unroll
  for (int t = 0; t < 4; ++t) acc[t] = v8f_zero();

  stage_B_async(woh, 512, col0, 0, bstage[0], tid);
  v16h a_cur = load_A_frag(aoh, 512, row0, 0, lane);

  for (int kk = 0; kk < 512; kk += 32) {
    const int  cur      = (kk >> 5) & 1;
    const bool has_next = (kk + 32) < 512;
    if (has_next)
      stage_B_async(woh, 512, col0, kk + 32, bstage[cur ^ 1], tid);
    if (has_next) wait_async_le2();
    else          wait_async_0();
    __syncthreads();

    const int kn = has_next ? kk + 32 : kk;
    const v16h a_nxt = load_A_frag(aoh, 512, row0, kn, lane);

    v16h b[4];
#pragma unroll
    for (int t = 0; t < 4; ++t)
      b[t] = load_Bt_frag(bstage[cur], 32, 0, 16 * t, lane);
#pragma unroll
    for (int t = 0; t < 4; ++t)
      acc[t] = wmma16(a_cur, b[t], acc[t]);

    __syncthreads();
    a_cur = a_nxt;
  }

  const int rbase = (lane < 16) ? 0 : 8;
#pragma unroll
  for (int t = 0; t < 4; ++t) {
    const int n = col0 + 16 * t + (lane & 15);
    const float bias = bo[n];
#pragma unroll
    for (int r = 0; r < 8; ++r) {
      const int m = row0 + rbase + r;
      out[(size_t)m * 4096 + n] = acc[t][r] + bias;
    }
  }
}

// ---------------------------------------------------------------------------
extern "C" void kernel_launch(void* const* d_in, const int* in_sizes, int n_in,
                              void* d_out, int out_size, void* d_ws, size_t ws_size,
                              hipStream_t stream) {
  (void)in_sizes; (void)n_in; (void)out_size; (void)ws_size;
  const float* x     = (const float*)d_in[0];
  const float* Wqk_w = (const float*)d_in[1];
  const float* Wqk_b = (const float*)d_in[2];
  const float* Wv_w  = (const float*)d_in[3];
  const float* Wv_b  = (const float*)d_in[4];
  const float* Wo_w  = (const float*)d_in[5];
  const float* Wo_b  = (const float*)d_in[6];
  float* out = (float*)d_out;

  char* ws = (char*)d_ws;
  size_t off = 0;
  auto walloc = [&](size_t bytes) -> void* {
    void* p = ws + off;
    off += (bytes + 255) & ~(size_t)255;
    return p;
  };

  _Float16* xh   = (_Float16*)walloc(8192ULL * 4096 * 2);   // 64 MB
  _Float16* wcat = (_Float16*)walloc(1536ULL * 4096 * 2);   // 12 MB (Wqk;Wv)
  _Float16* woh  = (_Float16*)walloc(4096ULL * 512 * 2);    // 4 MB
  _Float16* qh   = (_Float16*)walloc(8ULL * 8 * 1024 * 64 * 2);
  _Float16* kh   = (_Float16*)walloc(8ULL * 8 * 1024 * 64 * 2);
  _Float16* vth  = (_Float16*)walloc(8ULL * 8 * 64 * 1024 * 2);
  _Float16* aoh  = (_Float16*)walloc(8192ULL * 512 * 2);

  mha_pack_f16<<<4096, 256, 0, stream>>>(x, xh, 8192LL * 4096);
  mha_pack_f16<<<1024, 256, 0, stream>>>(Wqk_w, wcat, 1024LL * 4096);
  mha_pack_f16<<<512,  256, 0, stream>>>(Wv_w, wcat + 1024ULL * 4096, 512LL * 4096);
  mha_pack_f16<<<512,  256, 0, stream>>>(Wo_w, woh, 4096LL * 512);

  mha_proj_qkv<<<dim3(128, 24), 128, 0, stream>>>(xh, wcat, Wqk_b, Wv_b, qh, kh, vth);
  mha_attn<<<1024, 128, 0, stream>>>(qh, kh, vth, aoh);
  mha_out_proj<<<dim3(128, 64), 128, 0, stream>>>(aoh, woh, Wo_b, out);
}
